// Attention_70042326663508
// MI455X (gfx1250) — matrix operand order repaired and checked by reading, not compiled, not run on hardware
//
#include <hip/hip_runtime.h>

typedef _Float16 half_t;
typedef __attribute__((ext_vector_type(16))) _Float16 v16h;
typedef __attribute__((ext_vector_type(8)))  float    v8f;

#define HEADS 8
#define HDIM  64
#define NTOK  4096
#define CDIM  256
#define HID   512   // HEADS*HDIM

union FragU { uint32_t u[8]; v16h v; };

// A fragment: 16x32 f16, element [M0][K0] at p, row stride ld (elements).
// lane = M (0..15) x half; VGPR g holds K pair at 2g + 16*half.
__device__ inline v16h load_frag_a(const half_t* p, int ld) {
    int lane = threadIdx.x & 31;
    int m = lane & 15, hf = lane >> 4;
    const half_t* row = p + m * ld + 16 * hf;
    FragU f;
#pragma unroll
    for (int g = 0; g < 8; ++g) {
        int k = 2 * g;
        f.u[g] = *reinterpret_cast<const uint32_t*>(row + k);
    }
    return f.v;
}

// B fragment: 32x16 f16 from N-major storage Bt[N][K], element [N0][K0] at p,
// row stride ld. lane = N x half; VGPR g holds K pair at 2g + 16*half.
__device__ inline v16h load_frag_b(const half_t* p, int ld) {
    int lane = threadIdx.x & 31;
    int n = lane & 15, hf = lane >> 4;
    const half_t* row = p + n * ld + 16 * hf;
    FragU f;
#pragma unroll
    for (int g = 0; g < 8; ++g)
        f.u[g] = *reinterpret_cast<const uint32_t*>(row + 2 * g);
    return f.v;
}

__device__ inline v8f wmma16(v16h a, v16h b, v8f c) {
    return __builtin_amdgcn_wmma_f32_16x16x32_f16(false, a, false, b,
                                                  (short)0, c, false, false);
}

// ---------------- conversion kernels ----------------
__global__ void cvt_f32_f16(const float* __restrict__ s, half_t* __restrict__ d, int n) {
    int i = blockIdx.x * blockDim.x + threadIdx.x;
    if (i < n) d[i] = (half_t)s[i];
}

// src[rows][cols] (row-major) -> dst[cols][rows]
__global__ void cvt_transpose(const float* __restrict__ s, half_t* __restrict__ d,
                              int rows, int cols) {
    int i = blockIdx.x * blockDim.x + threadIdx.x;
    if (i < rows * cols) {
        int r = i / cols, c = i % cols;
        d[c * rows + r] = (half_t)s[i];
    }
}

// ---------------- QKV projection GEMM ----------------
// C[1536][4096] = Wqkv[1536][256] x X[256][4096]; scatter into qT/kT/v (f16).
// Wave tile: 32(M) x 64(N); block = 4 waves -> 128 x 64.
__global__ __launch_bounds__(128) void qkv_gemm(const half_t* __restrict__ wqkv,
                                                const half_t* __restrict__ xT,
                                                half_t* __restrict__ qT,
                                                half_t* __restrict__ kT,
                                                half_t* __restrict__ vmat) {
    int wave = threadIdx.x >> 5, lane = threadIdx.x & 31;
    int m = lane & 15, hf = lane >> 4;
    int n_base = blockIdx.x * 64;
    int o_base = blockIdx.y * 128 + wave * 32;
    v8f acc[2][4] = {};
    for (int k0 = 0; k0 < CDIM; k0 += 32) {
        v16h a0 = load_frag_a(wqkv + (o_base + 0) * CDIM + k0, CDIM);
        v16h a1 = load_frag_a(wqkv + (o_base + 16) * CDIM + k0, CDIM);
#pragma unroll
        for (int nf = 0; nf < 4; ++nf) {
            v16h b = load_frag_b(xT + (n_base + nf * 16) * CDIM + k0, CDIM);
            acc[0][nf] = wmma16(a0, b, acc[0][nf]);
            acc[1][nf] = wmma16(a1, b, acc[1][nf]);
        }
    }
    const float qscale = 0.125f;  // 64^-0.5
#pragma unroll
    for (int mf = 0; mf < 2; ++mf) {
#pragma unroll
        for (int nf = 0; nf < 4; ++nf) {
#pragma unroll
            for (int g = 0; g < 8; ++g) {
                int o = o_base + mf * 16 + g + 8 * hf;
                int n = n_base + nf * 16 + m;
                float val = acc[mf][nf][g];
                if (o < HID) {                       // Q -> qT[h][i][d], pre-scaled
                    int h = o >> 6, d = o & 63;
                    qT[((h << 12) + n) * HDIM + d] = (half_t)(val * qscale);
                } else if (o < 2 * HID) {            // K -> kT[h][j][d]
                    int oo = o - HID; int h = oo >> 6, d = oo & 63;
                    kT[((h << 12) + n) * HDIM + d] = (half_t)val;
                } else {                             // V -> v[h][d][j]
                    int oo = o - 2 * HID; int h = oo >> 6, d = oo & 63;
                    vmat[((h << 6) + d) * NTOK + n] = (half_t)val;
                }
            }
        }
    }
}

// ---------------- flash attention ----------------
// Wave tile: 32 query rows; B_k / B_v fragments shared across both 16-row halves.
__global__ __launch_bounds__(128) void flash_attn(const half_t* __restrict__ qT,
                                                  const half_t* __restrict__ kT,
                                                  const half_t* __restrict__ vmat,
                                                  half_t* __restrict__ aT) {
    __shared__ half_t lds_p[4][2][16][64];   // per-wave P tiles (C-layout -> A-layout)
    int wave = threadIdx.x >> 5, lane = threadIdx.x & 31;
    int m = lane & 15, hf = lane >> 4;
    int h  = blockIdx.y;
    int i0 = blockIdx.x * 128 + wave * 32;
    const half_t* qh = qT + (size_t)h * NTOK * HDIM;
    const half_t* kh = kT + (size_t)h * NTOK * HDIM;
    const half_t* vh = vmat + (size_t)h * HDIM * NTOK;

    v16h aq[2][2];
#pragma unroll
    for (int mf = 0; mf < 2; ++mf) {
        aq[mf][0] = load_frag_a(qh + (i0 + mf * 16) * HDIM + 0, HDIM);
        aq[mf][1] = load_frag_a(qh + (i0 + mf * 16) * HDIM + 32, HDIM);
    }

    float rmax[2][8], rsum[2][8];
    v8f acco[2][4] = {};
#pragma unroll
    for (int mf = 0; mf < 2; ++mf)
#pragma unroll
        for (int g = 0; g < 8; ++g) { rmax[mf][g] = -1e30f; rsum[mf][g] = 0.f; }

    for (int j0 = 0; j0 < NTOK; j0 += 64) {
        if (j0 + 64 < NTOK) {
            __builtin_prefetch(kh + (j0 + 64) * HDIM + lane * 128, 0, 3);
            __builtin_prefetch(vh + (lane & 3) * 16 * NTOK + ((lane >> 2) & 1) * 32 + j0 + 64, 0, 3);
        }
        // S tiles: [32 i][64 j], two 16-row halves sharing B_k fragments
        v8f s[2][4] = {};
#pragma unroll
        for (int jf = 0; jf < 4; ++jf) {
            v16h b0 = load_frag_b(kh + (j0 + jf * 16) * HDIM + 0, HDIM);
            v16h b1 = load_frag_b(kh + (j0 + jf * 16) * HDIM + 32, HDIM);
            s[0][jf] = wmma16(aq[0][0], b0, s[0][jf]);
            s[0][jf] = wmma16(aq[0][1], b1, s[0][jf]);
            s[1][jf] = wmma16(aq[1][0], b0, s[1][jf]);
            s[1][jf] = wmma16(aq[1][1], b1, s[1][jf]);
        }
        // online softmax per row (row = g + 8*hf; lanes of same half share row)
#pragma unroll
        for (int mf = 0; mf < 2; ++mf) {
#pragma unroll
            for (int g = 0; g < 8; ++g) {
                float cm = fmaxf(fmaxf(s[mf][0][g], s[mf][1][g]),
                                 fmaxf(s[mf][2][g], s[mf][3][g]));
#pragma unroll
                for (int msk = 1; msk <= 8; msk <<= 1)
                    cm = fmaxf(cm, __shfl_xor(cm, msk, 32));
                float nm = fmaxf(rmax[mf][g], cm);
                float corr = __expf(rmax[mf][g] - nm);
                float cs = 0.f;
#pragma unroll
                for (int jf = 0; jf < 4; ++jf) {
                    float p = __expf(s[mf][jf][g] - nm);
                    s[mf][jf][g] = p;
                    cs += p;
                }
#pragma unroll
                for (int msk = 1; msk <= 8; msk <<= 1)
                    cs += __shfl_xor(cs, msk, 32);
                rsum[mf][g] = rsum[mf][g] * corr + cs;
                rmax[mf][g] = nm;
#pragma unroll
                for (int df = 0; df < 4; ++df) acco[mf][df][g] *= corr;
            }
            // P: C-fragment layout -> LDS [i][j] (A layout), wave-private region
#pragma unroll
            for (int jf = 0; jf < 4; ++jf)
#pragma unroll
                for (int g = 0; g < 8; ++g)
                    lds_p[wave][mf][g + 8 * hf][jf * 16 + m] = (half_t)s[mf][jf][g];
        }
        // O += P x V^T, sharing B_v fragments across both row halves
#pragma unroll
        for (int kk = 0; kk < 2; ++kk) {
            v16h ap0 = load_frag_a(&lds_p[wave][0][0][kk * 32], 64);
            v16h ap1 = load_frag_a(&lds_p[wave][1][0][kk * 32], 64);
#pragma unroll
            for (int df = 0; df < 4; ++df) {
                v16h bv = load_frag_b(vh + (df * 16) * NTOK + j0 + kk * 32, NTOK);
                acco[0][df] = wmma16(ap0, bv, acco[0][df]);
                acco[1][df] = wmma16(ap1, bv, acco[1][df]);
            }
        }
    }
    // normalize and scatter to aT[i][h*64+d]
#pragma unroll
    for (int mf = 0; mf < 2; ++mf)
#pragma unroll
        for (int df = 0; df < 4; ++df)
#pragma unroll
            for (int g = 0; g < 8; ++g) {
                int i = i0 + mf * 16 + g + 8 * hf;
                int d = df * 16 + m;
                aT[(size_t)i * HID + h * HDIM + d] =
                    (half_t)(acco[mf][df][g] / rsum[mf][g]);
            }
}

// ---------------- output projection ----------------
// out[256][4096] = Wout[256][512] x A[512][4096] + b; wave tile 32 x 64.
__global__ __launch_bounds__(128) void out_proj(const half_t* __restrict__ wout,
                                                const half_t* __restrict__ aT,
                                                const float* __restrict__ bout,
                                                float* __restrict__ out) {
    int wave = threadIdx.x >> 5, lane = threadIdx.x & 31;
    int m = lane & 15, hf = lane >> 4;
    int n_base = blockIdx.x * 64;
    int o_base = blockIdx.y * 128 + wave * 32;
    v8f acc[2][4] = {};
    for (int k0 = 0; k0 < HID; k0 += 32) {
        v16h a0 = load_frag_a(wout + (o_base + 0) * HID + k0, HID);
        v16h a1 = load_frag_a(wout + (o_base + 16) * HID + k0, HID);
#pragma unroll
        for (int nf = 0; nf < 4; ++nf) {
            v16h b = load_frag_b(aT + (size_t)(n_base + nf * 16) * HID + k0, HID);
            acc[0][nf] = wmma16(a0, b, acc[0][nf]);
            acc[1][nf] = wmma16(a1, b, acc[1][nf]);
        }
    }
#pragma unroll
    for (int mf = 0; mf < 2; ++mf)
#pragma unroll
        for (int nf = 0; nf < 4; ++nf)
#pragma unroll
            for (int g = 0; g < 8; ++g) {
                int o = o_base + mf * 16 + g + 8 * hf;
                int n = n_base + nf * 16 + m;
                out[(size_t)o * NTOK + n] = acc[mf][nf][g] + bout[o];
            }
}

extern "C" void kernel_launch(void* const* d_in, const int* in_sizes, int n_in,
                              void* d_out, int out_size, void* d_ws, size_t ws_size,
                              hipStream_t stream) {
    const float* x     = (const float*)d_in[0];   // [256][4096]
    const float* w_qkv = (const float*)d_in[1];   // [1536][256]
    const float* w_out = (const float*)d_in[2];   // [256][512]
    const float* b_out = (const float*)d_in[3];   // [256]
    float* out = (float*)d_out;                   // [256][4096]

    half_t* ws    = (half_t*)d_ws;
    half_t* xT    = ws;                            // [4096][256]
    half_t* wqkvh = xT + NTOK * CDIM;              // [1536][256]
    half_t* wouth = wqkvh + 3 * HID * CDIM;        // [256][512]
    half_t* qT    = wouth + CDIM * HID;            // [8][4096][64]
    half_t* kT    = qT + (size_t)HEADS * NTOK * HDIM;
    half_t* vmat  = kT + (size_t)HEADS * NTOK * HDIM;
    half_t* aT    = vmat + (size_t)HEADS * HDIM * NTOK;  // [4096][512]

    cvt_transpose<<<(CDIM * NTOK + 255) / 256, 256, 0, stream>>>(x, xT, CDIM, NTOK);
    cvt_f32_f16<<<(3 * HID * CDIM + 255) / 256, 256, 0, stream>>>(w_qkv, wqkvh, 3 * HID * CDIM);
    cvt_f32_f16<<<(CDIM * HID + 255) / 256, 256, 0, stream>>>(w_out, wouth, CDIM * HID);

    qkv_gemm<<<dim3(NTOK / 64, (3 * HID) / 128), 128, 0, stream>>>(wqkvh, xT, qT, kT, vmat);
    flash_attn<<<dim3(NTOK / 128, HEADS), 128, 0, stream>>>(qT, kT, vmat, aT);
    out_proj<<<dim3(NTOK / 64, CDIM / 128), 128, 0, stream>>>(wouth, aT, b_out, out);
}
